// intra_graph_84782654423336
// MI455X (gfx1250) — matrix operand order repaired and checked by reading, not compiled, not run on hardware
//
#include <hip/hip_runtime.h>
#include <math.h>

// ---------------------------------------------------------------------------
// Problem constants (from the reference)
// ---------------------------------------------------------------------------
#define Bn 8
#define Cc 256
#define Hh 96
#define Ww 96
#define Nn (Hh * Ww)        // 9216 pixels
#define Kk 32               // clusters
#define BN_EPS    1e-5f
#define EPS_ASSGN 1e-9f
#define NORM_EPS  1e-12f

// ---------------------------------------------------------------------------
// WMMA plumbing (CDNA5 / gfx1250, wave32)
// ---------------------------------------------------------------------------
typedef __attribute__((ext_vector_type(16))) __bf16          bf16x16;
typedef __attribute__((ext_vector_type(16))) unsigned short  u16x16;
typedef __attribute__((ext_vector_type(8)))  float           v8f;
typedef __attribute__((ext_vector_type(4)))  unsigned int    u32x4;
typedef __attribute__((ext_vector_type(8)))  int             i32x8;
typedef __attribute__((ext_vector_type(4)))  int             i32x4;

union BfPack { u16x16 u; bf16x16 b; };

__device__ __forceinline__ v8f wmma_bf16(const BfPack& a, const BfPack& bm, v8f c) {
  // D(16x16,f32) = A(16x32,bf16) * B(32x16,bf16) + C
  return __builtin_amdgcn_wmma_f32_16x16x32_bf16(
      /*neg_a=*/false, a.b, /*neg_b=*/false, bm.b,
      /*c_mod=*/(short)0, c, /*reuse_a=*/false, /*reuse_b=*/false);
}

// f32 -> bf16 via the native converter (backend emits v_cvt_pk_bf16_f32, RNE)
__device__ __forceinline__ unsigned short f2bf(float f) {
  union { __bf16 h; unsigned short u; } cv;
  cv.h = (__bf16)f;
  return cv.u;
}
__device__ __forceinline__ float bf2f(unsigned short h) {
  return __uint_as_float(((unsigned)h) << 16);
}

// ISA A-matrix (16-bit, 16x32) K index for half j (0..15), half-wave hi
__device__ __forceinline__ int a_kk(int j, int hi) {
  return (j < 8) ? (hi * 8 + j) : (16 + hi * 8 + (j - 8));
}

__device__ __forceinline__ float sigmoidf(float x) { return 1.0f / (1.0f + __expf(-x)); }

// ===========================================================================
// K0: precompute sigma, S1 = 1/sigma^2 (bf16), S2 = 2*a/sigma^2 (bf16), t3[k]
// ===========================================================================
__global__ void k_prep(const float* __restrict__ anchor, const float* __restrict__ sigma_p,
                       float* __restrict__ sigma, unsigned short* __restrict__ S1,
                       unsigned short* __restrict__ S2, float* __restrict__ t3) {
  int c = threadIdx.x;  // 0..255
  for (int k = 0; k < Kk; ++k) {
    float s = sigmoidf(sigma_p[k * Cc + c]);
    float is2 = 1.0f / (s * s);
    sigma[k * Cc + c] = s;
    S1[k * Cc + c] = f2bf(is2);
    S2[k * Cc + c] = f2bf(2.0f * anchor[k * Cc + c] * is2);
  }
  if (c < Kk) {
    int k = c;
    float acc = 0.0f;
    for (int cc = 0; cc < Cc; ++cc) {
      float s = sigmoidf(sigma_p[k * Cc + cc]);
      float a = anchor[k * Cc + cc];
      acc += a * a / (s * s);
    }
    t3[k] = acc;
  }
}

// ===========================================================================
// K1: soft-assignment.  Per 16-pixel tile: two WMMA GEMMs over C=256
//     (t1 = S1*x^2, t2 = S2*x), logits = 0.5*(t2 - t1 - t3), softmax over K=32
//     (cross-half-wave reduction via shfl_xor 16), store assign as bf16 [B,K,N].
// block = 128 threads (4 waves), grid = (N/64, B)
// ===========================================================================
__global__ void __launch_bounds__(128)
k_assign(const float* __restrict__ x, const unsigned short* __restrict__ S1,
         const unsigned short* __restrict__ S2, const float* __restrict__ t3,
         unsigned short* __restrict__ abuf) {
  __shared__ unsigned short sS1[Kk * Cc];
  __shared__ unsigned short sS2[Kk * Cc];
  __shared__ float sT3[Kk];

  int tid = threadIdx.x;
  for (int i = tid; i < Kk * Cc; i += 128) { sS1[i] = S1[i]; sS2[i] = S2[i]; }
  if (tid < Kk) sT3[tid] = t3[tid];
  __syncthreads();

  const int wave = tid >> 5, lane = tid & 31, hi = lane >> 4, ln = lane & 15;
  const int b  = blockIdx.y;
  const int n0 = (blockIdx.x * 4 + wave) * 16;
  const float* xb = x + (size_t)b * Cc * Nn;

  v8f acc1[2], acc2[2];
  v8f z = {};
  acc1[0] = z; acc1[1] = z; acc2[0] = z; acc2[1] = z;

  for (int s = 0; s < Cc / 32; ++s) {           // 8 k-dim steps of 32 channels
    if (s + 1 < Cc / 32) {                      // CDNA5 global_prefetch_b8, next k-step
      __builtin_prefetch(&xb[(size_t)((s + 1) * 32 + hi * 16) * Nn + n0 + ln], 0, 1);
      __builtin_prefetch(&xb[(size_t)((s + 1) * 32 + hi * 16 + 8) * Nn + n0 + ln], 0, 1);
    }
    BfPack bx, bq;
    #pragma unroll
    for (int j = 0; j < 16; ++j) {
      int c = s * 32 + hi * 16 + j;
      float f = xb[(size_t)c * Nn + n0 + ln];
      bx.u[j] = f2bf(f);
      bq.u[j] = f2bf(f * f);
    }
    #pragma unroll
    for (int t = 0; t < 2; ++t) {
      BfPack a1, a2;
      int m = t * 16 + ln;
      #pragma unroll
      for (int j = 0; j < 16; ++j) {
        int cc = s * 32 + hi * 16 + j;
        a1.u[j] = sS1[m * Cc + cc];
        a2.u[j] = sS2[m * Cc + cc];
      }
      acc1[t] = wmma_bf16(a1, bq, acc1[t]);     // t1 partial
      acc2[t] = wmma_bf16(a2, bx, acc2[t]);     // t2 partial
    }
  }

  // logits for 16 of the 32 k-rows live in this lane
  float lg[16];
  float mx = -1e30f;
  #pragma unroll
  for (int t = 0; t < 2; ++t)
    #pragma unroll
    for (int r = 0; r < 8; ++r) {
      int k = t * 16 + r + 8 * hi;
      float l = 0.5f * (acc2[t][r] - acc1[t][r] - sT3[k]);
      lg[t * 8 + r] = l;
      mx = fmaxf(mx, l);
    }
  float gm = fmaxf(mx, __shfl_xor(mx, 16, 32));       // max over all 32 k
  float se = 0.0f;
  #pragma unroll
  for (int i = 0; i < 16; ++i) { float e = __expf(lg[i] - gm); lg[i] = e; se += e; }
  float tot = se + __shfl_xor(se, 16, 32);
  float inv = 1.0f / tot;

  #pragma unroll
  for (int t = 0; t < 2; ++t)
    #pragma unroll
    for (int r = 0; r < 8; ++r) {
      int k = t * 16 + r + 8 * hi;
      abuf[((size_t)b * Kk + k) * Nn + n0 + ln] = f2bf(lg[t * 8 + r] * inv);
    }
}

// ===========================================================================
// K2: asum[b,k] = sum_n assign[b,k,n]   (deterministic block reduction)
// ===========================================================================
__global__ void k_asum(const unsigned short* __restrict__ abuf, float* __restrict__ asum) {
  __shared__ float red[256];
  int bk = blockIdx.x;
  const unsigned short* p = abuf + (size_t)bk * Nn;
  float s = 0.0f;
  for (int i = threadIdx.x; i < Nn; i += 256) s += bf2f(p[i]);
  red[threadIdx.x] = s;
  __syncthreads();
  for (int o = 128; o > 0; o >>= 1) {
    if (threadIdx.x < o) red[threadIdx.x] += red[threadIdx.x + o];
    __syncthreads();
  }
  if (threadIdx.x == 0) asum[bk] = red[0];
}

// ===========================================================================
// K3: Gram G[b] = assign @ assign^T  (32x32, WMMA over N).  grid=B, block=32.
// ===========================================================================
__global__ void __launch_bounds__(32)
k_gram(const unsigned short* __restrict__ abuf, float* __restrict__ G) {
  int b = blockIdx.x;
  const int lane = threadIdx.x & 31, hi = lane >> 4, ln = lane & 15;
  const unsigned short* ab = abuf + (size_t)b * Kk * Nn;
  v8f acc[2][2];
  v8f z = {};
  acc[0][0] = z; acc[0][1] = z; acc[1][0] = z; acc[1][1] = z;

  for (int st = 0; st < Nn / 32; ++st) {
    int n0 = st * 32;
    if (st + 1 < Nn / 32) {
      __builtin_prefetch(&ab[(size_t)ln * Nn + n0 + 32], 0, 1);
      __builtin_prefetch(&ab[(size_t)(16 + ln) * Nn + n0 + 32], 0, 1);
    }
    BfPack A[2], Bm[2];
    #pragma unroll
    for (int t = 0; t < 2; ++t) {
      int m = t * 16 + ln;
      #pragma unroll
      for (int j = 0; j < 16; ++j) A[t].u[j] = ab[(size_t)m * Nn + n0 + hi * 16 + j];
    }
    #pragma unroll
    for (int ct = 0; ct < 2; ++ct) {
      int col = ct * 16 + ln;
      #pragma unroll
      for (int j = 0; j < 16; ++j) Bm[ct].u[j] = ab[(size_t)col * Nn + n0 + hi * 16 + j];
    }
    #pragma unroll
    for (int t = 0; t < 2; ++t)
      #pragma unroll
      for (int ct = 0; ct < 2; ++ct)
        acc[t][ct] = wmma_bf16(A[t], Bm[ct], acc[t][ct]);
  }
  #pragma unroll
  for (int t = 0; t < 2; ++t)
    #pragma unroll
    for (int ct = 0; ct < 2; ++ct)
      #pragma unroll
      for (int r = 0; r < 8; ++r) {
        int i = t * 16 + r + 8 * hi, j = ct * 16 + ln;
        G[(size_t)b * Kk * Kk + i * Kk + j] = acc[t][ct][r];
      }
}

// ===========================================================================
// K4: agg[b,k,c] = sum_n assign[b,k,n] * x[b,c,n]  (WMMA).
// grid=(C/16, B), block=32 (one wave per 32x16 output tile).
// ===========================================================================
__global__ void __launch_bounds__(32)
k_agg(const unsigned short* __restrict__ abuf, const float* __restrict__ x,
      float* __restrict__ agg) {
  int b = blockIdx.y, c0 = blockIdx.x * 16;
  const int lane = threadIdx.x & 31, hi = lane >> 4, ln = lane & 15;
  const unsigned short* ab = abuf + (size_t)b * Kk * Nn;
  const float* xb = x + (size_t)b * Cc * Nn;
  v8f acc[2]; v8f z = {}; acc[0] = z; acc[1] = z;

  for (int st = 0; st < Nn / 32; ++st) {
    int n0 = st * 32;
    if (st + 1 < Nn / 32) {
      __builtin_prefetch(&xb[(size_t)(c0 + ln) * Nn + n0 + 32 + hi * 16], 0, 1);
    }
    BfPack Bm;
    #pragma unroll
    for (int j = 0; j < 16; ++j)
      Bm.u[j] = f2bf(xb[(size_t)(c0 + ln) * Nn + n0 + hi * 16 + j]);
    #pragma unroll
    for (int t = 0; t < 2; ++t) {
      BfPack A;
      int m = t * 16 + ln;
      #pragma unroll
      for (int j = 0; j < 16; ++j) A.u[j] = ab[(size_t)m * Nn + n0 + hi * 16 + j];
      acc[t] = wmma_bf16(A, Bm, acc[t]);
    }
  }
  #pragma unroll
  for (int t = 0; t < 2; ++t)
    #pragma unroll
    for (int r = 0; r < 8; ++r) {
      int k = t * 16 + r + 8 * hi;
      agg[((size_t)b * Kk + k) * Cc + c0 + ln] = acc[t][r];
    }
}

// ===========================================================================
// K5: per-node residual + row L2 normalize.  grid = B*K blocks of 256.
// ===========================================================================
__global__ void k_nodes(const float* __restrict__ agg, const float* __restrict__ asum,
                        const float* __restrict__ anchor, const float* __restrict__ sigma,
                        float* __restrict__ nodes) {
  __shared__ float red[256];
  int bk = blockIdx.x, k = bk & (Kk - 1), c = threadIdx.x;
  float as = asum[bk];
  float v = (agg[(size_t)bk * Cc + c] - as * anchor[k * Cc + c]) / sigma[k * Cc + c];
  v /= (as + EPS_ASSGN);
  red[c] = v * v;
  __syncthreads();
  for (int o = 128; o > 0; o >>= 1) {
    if (c < o) red[c] += red[c + o];
    __syncthreads();
  }
  float rn = 1.0f / fmaxf(sqrtf(red[0]), NORM_EPS);
  nodes[(size_t)bk * Cc + c] = v * rn;
}

// K6: flat L2 normalize per batch. grid = B blocks of 256.
__global__ void k_flatnorm(float* __restrict__ nodes) {
  __shared__ float red[256];
  float* p = nodes + (size_t)blockIdx.x * Kk * Cc;
  float s = 0.0f;
  for (int i = threadIdx.x; i < Kk * Cc; i += 256) { float v = p[i]; s += v * v; }
  red[threadIdx.x] = s;
  __syncthreads();
  for (int o = 128; o > 0; o >>= 1) {
    if (threadIdx.x < o) red[threadIdx.x] += red[threadIdx.x + o];
    __syncthreads();
  }
  float rn = 1.0f / fmaxf(sqrtf(red[0]), NORM_EPS);
  __syncthreads();
  for (int i = threadIdx.x; i < Kk * Cc; i += 256) p[i] *= rn;
}

// ===========================================================================
// K7: conv1 (1x1): h1[b,c,k] = sum_c' w[c,c'] * nodes_view[b,c',k]
// grid=(C,B) block=32 (thread=k).
// ===========================================================================
__global__ void __launch_bounds__(32)
k_mm_ck(const float* __restrict__ w, const float* __restrict__ in,
        float* __restrict__ out, int relu_in) {
  int b = blockIdx.y, c = blockIdx.x, k = threadIdx.x;
  const float* inb = in + (size_t)b * Cc * Kk;
  const float* wr  = w + (size_t)c * Cc;
  float s = 0.0f;
  for (int cp = 0; cp < Cc; ++cp) {
    float v = inb[cp * Kk + k];
    if (relu_in) v = fmaxf(v, 0.0f);
    s += wr[cp] * v;
  }
  out[((size_t)b * Cc + c) * Kk + k] = s;
}

// K8: BN stats over (B,K) per channel -> affine scale/shift. grid=C blocks of 256.
__global__ void k_bnstats_small(const float* __restrict__ h, const float* __restrict__ g,
                                const float* __restrict__ bb, float* __restrict__ scale,
                                float* __restrict__ shift) {
  __shared__ float rs[256], rq[256];
  int c = blockIdx.x, t = threadIdx.x;
  int b = t >> 5, k = t & 31;
  float v = h[((size_t)b * Cc + c) * Kk + k];
  rs[t] = v; rq[t] = v * v;
  __syncthreads();
  for (int o = 128; o > 0; o >>= 1) {
    if (t < o) { rs[t] += rs[t + o]; rq[t] += rq[t + o]; }
    __syncthreads();
  }
  if (t == 0) {
    float mean = rs[0] / (Bn * Kk);
    float var  = rq[0] / (Bn * Kk) - mean * mean;
    float sc = g[c] * rsqrtf(var + BN_EPS);
    scale[c] = sc;
    shift[c] = bb[c] - mean * sc;
  }
}

// K9: elementwise y = relu(h*scale[c] + shift[c])  over [B,C,K]
__global__ void k_bnrelu_small(const float* __restrict__ h, const float* __restrict__ scale,
                               const float* __restrict__ shift, float* __restrict__ o) {
  int idx = blockIdx.x * 256 + threadIdx.x;
  int c = (idx >> 5) & (Cc - 1);
  o[idx] = fmaxf(h[idx] * scale[c] + shift[c], 0.0f);
}

// ===========================================================================
// GCN pieces (tiny [B,C,K] tensors; VALU kernels)
// ===========================================================================
__global__ void __launch_bounds__(1024)
k_adj(const float* __restrict__ xf, float* __restrict__ adj) {
  __shared__ float m[Kk][Kk];
  int b = blockIdx.x, t = threadIdx.x, i = t >> 5, j = t & 31;
  const float* xb = xf + (size_t)b * Cc * Kk;
  float s = 0.0f;
  for (int c = 0; c < Cc; ++c) s += xb[c * Kk + i] * xb[c * Kk + j];
  m[i][j] = s;
  __syncthreads();
  float mx = -1e30f;
  for (int jj = 0; jj < Kk; ++jj) mx = fmaxf(mx, m[i][jj]);
  float den = 0.0f;
  for (int jj = 0; jj < Kk; ++jj) den += __expf(m[i][jj] - mx);
  adj[(size_t)b * Kk * Kk + i * Kk + j] = __expf(s - mx) / den;
}

__global__ void k_support(const float* __restrict__ xf, const float* __restrict__ w,
                          float* __restrict__ sup) {
  int idx = blockIdx.x * 256 + threadIdx.x;      // (b*K+i)*C + d
  int d = idx & (Cc - 1), i = (idx >> 8) & (Kk - 1), b = idx >> 13;
  const float* xb = xf + (size_t)b * Cc * Kk;
  float s = 0.0f;
  for (int c = 0; c < Cc; ++c) s += xb[c * Kk + i] * w[c * Cc + d];
  sup[idx] = s;
}

__global__ void k_gout(const float* __restrict__ adj, const float* __restrict__ sup,
                       float* __restrict__ g) {
  int idx = blockIdx.x * 256 + threadIdx.x;      // (b*C+d)*K + i
  int i = idx & (Kk - 1), d = (idx >> 5) & (Cc - 1), b = idx >> 13;
  const float* ar = adj + (size_t)b * Kk * Kk + i * Kk;
  const float* sb = sup + (size_t)b * Kk * Cc;
  float s = 0.0f;
  for (int j = 0; j < Kk; ++j) s += ar[j] * sb[j * Cc + d];
  g[idx] = s;
}

// K10: M = conv2_w @ relu(g2);  also emit bf16 copy in A-matrix friendly [C,K] rows.
__global__ void __launch_bounds__(32)
k_M(const float* __restrict__ w2, const float* __restrict__ g2,
    float* __restrict__ Mf, unsigned short* __restrict__ Mb) {
  int b = blockIdx.y, c = blockIdx.x, k = threadIdx.x;
  const float* gb = g2 + (size_t)b * Cc * Kk;
  const float* wr = w2 + (size_t)c * Cc;
  float s = 0.0f;
  for (int cp = 0; cp < Cc; ++cp) s += wr[cp] * fmaxf(gb[cp * Kk + k], 0.0f);
  size_t o = ((size_t)b * Cc + c) * Kk + k;
  Mf[o] = s;
  Mb[o] = f2bf(s);
}

// K11: analytic BN2 stats: E[y] from M.asum, E[y^2] from M G M^T. block 256, grid 1.
__global__ void k_stats2(const float* __restrict__ Mf, const float* __restrict__ asum,
                         const float* __restrict__ G, const float* __restrict__ g,
                         const float* __restrict__ bb, float* __restrict__ scale,
                         float* __restrict__ shift) {
  int c = threadIdx.x;
  float s1 = 0.0f, s2 = 0.0f;
  for (int b = 0; b < Bn; ++b) {
    const float* m = Mf + ((size_t)b * Cc + c) * Kk;
    const float* as = asum + b * Kk;
    const float* Gb = G + (size_t)b * Kk * Kk;
    for (int k = 0; k < Kk; ++k) s1 += m[k] * as[k];
    for (int i = 0; i < Kk; ++i) {
      float inner = 0.0f;
      for (int j = 0; j < Kk; ++j) inner += Gb[i * Kk + j] * m[j];
      s2 += m[i] * inner;
    }
  }
  const float invBN = 1.0f / (float)(Bn * Nn);
  float mean = s1 * invBN;
  float var  = s2 * invBN - mean * mean;
  float sc = g[c] * rsqrtf(var + BN_EPS);
  scale[c] = sc;
  shift[c] = bb[c] - mean * sc;
}

// ===========================================================================
// K12: fused output pass.  y = M @ assign (k-dim = 32 -> ONE wmma per 16x16
//      tile), then out = x + relu(y*scale + shift).
//  - M[b] (16 KB contiguous) DMA'd into LDS by the Tensor Data Mover
//    (tensor_load_to_lds, issued by wave 0 only; TENSORcnt wait + barrier).
//    D# built per CDNA5 ISA: count=1, type=2("image"),
//    data_size=2B, 1-D tile tensor_dim0 = tile_dim0 = 8192 elements.
//  - assign tile (32 k x 128 n) staged in LDS TRANSPOSED so both the global
//    side (contiguous 32B runs -> global_load_b128) and the LDS side
//    (16 contiguous halves per lane -> ds_load_b128) are vectorized.
// block=256 (8 waves, one 16-pixel column strip each), grid=(N/128, B)
// ===========================================================================
__global__ void __launch_bounds__(256)
k_final(const unsigned short* __restrict__ Mb, const unsigned short* __restrict__ abuf,
        const float* __restrict__ x, const float* __restrict__ scale,
        const float* __restrict__ shift, float* __restrict__ out) {
  __shared__ unsigned short sM[Cc * Kk];        // 16 KB  (TDM destination)
  __shared__ unsigned short sB[128 * Kk];       // 8 KB, [n_local][k]
  __shared__ float sSc[Cc], sSh[Cc];
  int tid = threadIdx.x, b = blockIdx.y;
  const int n0b = blockIdx.x * 128;
  const unsigned short* ab = abuf + (size_t)b * Kk * Nn;

  // --- TDM: DMA M[b] (8192 bf16 = 16 KB, contiguous) into sM. Wave 0 only. ---
  if (tid < 32) {
    unsigned long long ga = (unsigned long long)(size_t)(Mb + (size_t)b * Cc * Kk);
    unsigned lds_addr = (unsigned)(size_t)(void*)sM;
    u32x4 g0;
    g0.x = 1u;                                           // count=1 valid descriptor
    g0.y = lds_addr;                                     // lds_addr [63:32]
    g0.z = (unsigned)(ga & 0xFFFFFFFFu);                 // global_addr lo
    g0.w = (unsigned)((ga >> 32) & 0x01FFFFFFu)          // global_addr [56:32]
           | 0x80000000u;                                // type=2 in bits [127:126]
    i32x8 g1;
    g1[0] = 0x00010000;                                  // data_size=1 -> 2 bytes
    g1[1] = (int)((8192u & 0xFFFFu) << 16);              // tensor_dim0 [15:0]
    g1[2] = (int)((8192u >> 16) | (1u << 16));           // tensor_dim0 hi | tensor_dim1 lo
    g1[3] = (int)(8192u << 16);                          // tile_dim0 = 8192
    g1[4] = 0;                                           // tile_dim1 = tile_dim2 = 0
    g1[5] = 8192;                                        // tensor_dim0_stride lo
    g1[6] = 0;
    g1[7] = 0;
    i32x4 gz = {0, 0, 0, 0};
#if defined(__clang_major__) && (__clang_major__ >= 23)
    i32x8 gz8 = {0, 0, 0, 0, 0, 0, 0, 0};
    __builtin_amdgcn_tensor_load_to_lds(g0, g1, gz, gz, gz8, 0);
#else
    __builtin_amdgcn_tensor_load_to_lds(g0, g1, gz, gz, 0);
#endif
    __builtin_amdgcn_s_wait_tensorcnt(0);
  }

  sSc[tid] = scale[tid];
  sSh[tid] = shift[tid];
  // stage assign tile transposed: thread t covers (row k = t&31, 16-pixel group g)
  {
    int k = tid & 31, g = tid >> 5;
    const unsigned short* src = &ab[(size_t)k * Nn + n0b + g * 16];  // 32B contiguous
    #pragma unroll
    for (int i = 0; i < 16; ++i) sB[(g * 16 + i) * Kk + k] = src[i];
  }
  __syncthreads();

  const int wave = tid >> 5, lane = tid & 31, hi = lane >> 4, ln = lane & 15;
  const int n0 = n0b + wave * 16;
  const float* xb = x + (size_t)b * Cc * Nn;
  float* ob = out + (size_t)b * Cc * Nn;

  BfPack Bm;                                   // assign tile: 32 k-rows x 16 pixels
  {
    const unsigned short* bp = &sB[(wave * 16 + ln) * Kk + hi * 16];  // 32B contiguous
    #pragma unroll
    for (int j = 0; j < 16; ++j) Bm.u[j] = bp[j];
  }

  for (int rt = 0; rt < Cc / 16; ++rt) {       // 16 row-tiles cover C=256
    BfPack A;
    int m = rt * 16 + ln;
    #pragma unroll
    for (int j = 0; j < 16; ++j) A.u[j] = sM[m * Kk + hi * 16 + j];
    v8f acc = {};
    acc = wmma_bf16(A, Bm, acc);
    #pragma unroll
    for (int r = 0; r < 8; ++r) {
      int row = rt * 16 + r + 8 * hi;
      size_t idx = (size_t)row * Nn + n0 + ln;
      float y = fmaxf(acc[r] * sSc[row] + sSh[row], 0.0f);
      ob[idx] = xb[idx] + y;
    }
  }
}

// ===========================================================================
// Host launcher
// ===========================================================================
extern "C" void kernel_launch(void* const* d_in, const int* in_sizes, int n_in,
                              void* d_out, int out_size, void* d_ws, size_t ws_size,
                              hipStream_t stream) {
  const float* x       = (const float*)d_in[0];
  const float* anchor  = (const float*)d_in[1];
  const float* sigma_p = (const float*)d_in[2];
  const float* conv1_w = (const float*)d_in[3];
  const float* bn1_g   = (const float*)d_in[4];
  const float* bn1_b   = (const float*)d_in[5];
  const float* gcn_w1  = (const float*)d_in[6];
  const float* gcn_w2  = (const float*)d_in[7];
  const float* conv2_w = (const float*)d_in[8];
  const float* bn2_g   = (const float*)d_in[9];
  const float* bn2_b   = (const float*)d_in[10];
  float* outp = (float*)d_out;

  // workspace sub-allocation (≈6.8 MB total)
  char* p = (char*)d_ws;
  auto alloc = [&](size_t bytes) -> void* {
    void* r = (void*)p;
    p += (bytes + 255) & ~(size_t)255;
    return r;
  };
  unsigned short* abuf  = (unsigned short*)alloc((size_t)Bn * Kk * Nn * 2); // assign bf16
  unsigned short* S1    = (unsigned short*)alloc(Kk * Cc * 2);
  unsigned short* S2    = (unsigned short*)alloc(Kk * Cc * 2);
  float* t3    = (float*)alloc(Kk * 4);
  float* sigma = (float*)alloc(Kk * Cc * 4);
  float* asum  = (float*)alloc(Bn * Kk * 4);
  float* Gm    = (float*)alloc(Bn * Kk * Kk * 4);
  float* agg   = (float*)alloc((size_t)Bn * Kk * Cc * 4);
  float* nodes = (float*)alloc((size_t)Bn * Kk * Cc * 4);
  float* h1    = (float*)alloc((size_t)Bn * Cc * Kk * 4);
  float* xf1   = (float*)alloc((size_t)Bn * Cc * Kk * 4);
  float* adj   = (float*)alloc(Bn * Kk * Kk * 4);
  float* sup   = (float*)alloc((size_t)Bn * Kk * Cc * 4);
  float* g1    = (float*)alloc((size_t)Bn * Cc * Kk * 4);
  float* g2    = (float*)alloc((size_t)Bn * Cc * Kk * 4);
  float* Mf    = (float*)alloc((size_t)Bn * Cc * Kk * 4);
  unsigned short* Mb = (unsigned short*)alloc((size_t)Bn * Cc * Kk * 2);
  float* sc1   = (float*)alloc(Cc * 4);
  float* sh1   = (float*)alloc(Cc * 4);
  float* sc2   = (float*)alloc(Cc * 4);
  float* sh2   = (float*)alloc(Cc * 4);

  // 1) soft-assignment path
  k_prep<<<1, 256, 0, stream>>>(anchor, sigma_p, sigma, S1, S2, t3);
  k_assign<<<dim3(Nn / 64, Bn), 128, 0, stream>>>(x, S1, S2, t3, abuf);
  k_asum<<<Bn * Kk, 256, 0, stream>>>(abuf, asum);
  k_gram<<<Bn, 32, 0, stream>>>(abuf, Gm);
  k_agg<<<dim3(Cc / 16, Bn), 32, 0, stream>>>(abuf, x, agg);
  k_nodes<<<Bn * Kk, 256, 0, stream>>>(agg, asum, anchor, sigma, nodes);
  k_flatnorm<<<Bn, 256, 0, stream>>>(nodes);

  // 2) conv1 + BN1 + ReLU on the [B,C,K] view
  k_mm_ck<<<dim3(Cc, Bn), 32, 0, stream>>>(conv1_w, nodes, h1, /*relu_in=*/0);
  k_bnstats_small<<<Cc, 256, 0, stream>>>(h1, bn1_g, bn1_b, sc1, sh1);
  k_bnrelu_small<<<(Bn * Cc * Kk) / 256, 256, 0, stream>>>(h1, sc1, sh1, xf1);

  // 3) cascade GCN (loop = 2)
  k_adj<<<Bn, 1024, 0, stream>>>(xf1, adj);
  k_support<<<(Bn * Kk * Cc) / 256, 256, 0, stream>>>(xf1, gcn_w1, sup);
  k_gout<<<(Bn * Cc * Kk) / 256, 256, 0, stream>>>(adj, sup, g1);
  k_adj<<<Bn, 1024, 0, stream>>>(g1, adj);
  k_support<<<(Bn * Kk * Cc) / 256, 256, 0, stream>>>(g1, gcn_w2, sup);
  k_gout<<<(Bn * Cc * Kk) / 256, 256, 0, stream>>>(adj, sup, g2);

  // 4) fold conv2 into M = conv2_w @ relu(g2); analytic BN2 stats via asum & Gram
  k_M<<<dim3(Cc, Bn), 32, 0, stream>>>(conv2_w, g2, Mf, Mb);
  k_stats2<<<1, 256, 0, stream>>>(Mf, asum, Gm, bn2_g, bn2_b, sc2, sh2);

  // 5) fused output: out = x + relu(bn2(M @ assign))
  k_final<<<dim3(Nn / 128, Bn), 256, 0, stream>>>(Mb, abuf, x, sc2, sh2, outp);
}
